// GridHead_54022098649983
// MI455X (gfx1250) — compile-verified, matrix operand order checked by reading, not run on hardware
//
#include <hip/hip_runtime.h>
#include <hip/hip_bf16.h>

// ---------------- types ----------------
typedef __bf16 bf16x8  __attribute__((ext_vector_type(8)));
typedef __bf16 bf16x16 __attribute__((ext_vector_type(16)));
typedef float  f32x8   __attribute__((ext_vector_type(8)));

static __device__ __forceinline__ unsigned short f32_to_bf16(float f) {
    union { float f; unsigned int u; } v; v.f = f;
    unsigned int u = v.u;
    unsigned int r = (u + 0x7FFFu + ((u >> 16) & 1u)) >> 16;
    return (unsigned short)r;
}

// ---------------- problem constants ----------------
#define BATCH 64
#define CIN   256
#define CMID  576
#define HIN   14
#define HMID  7
#define PMID  49            // 7*7
#define NGRID 9
#define SPC   64
#define NEDGE 24
#define M_MID (BATCH*PMID)  // 3136
#define MPAD  3200          // M_MID padded to 128-row GEMM tiles (25 tiles)

__constant__ int c_edge_src[NEDGE] = {1,3, 0,2,4, 1,5, 0,4,6, 1,3,5,7, 2,4,8, 3,7, 4,6,8, 5,7};
__constant__ int c_node_off[NGRID+1] = {0,2,5,7,10,14,17,19,22,24};

// ---------------- TDM (Tensor Data Mover) support probe ----------------
#if defined(__has_builtin)
#  if __has_builtin(__builtin_amdgcn_tensor_load_to_lds)
#    define USE_TDM 1
#  endif
#endif
#ifndef USE_TDM
#  define USE_TDM 0
#endif

#if USE_TDM
typedef unsigned int u32x4 __attribute__((ext_vector_type(4)));
typedef int i32x4 __attribute__((ext_vector_type(4)));
typedef int i32x8 __attribute__((ext_vector_type(8)));
typedef __attribute__((address_space(3))) char lds_char;

static __device__ __forceinline__ unsigned lds_off_u32(void* p) {
    return (unsigned)(unsigned long long)(lds_char*)p;
}

// Issue a 2D bf16 tile DMA: tile_k contiguous elements per row, tile_rows rows,
// global row stride = row_stride_elems, packed row-major into LDS at lds_addr.
static __device__ __forceinline__ void tdm_load_tile_bf16(
        unsigned lds_addr, const unsigned short* gptr,
        unsigned tile_k, unsigned tile_rows, unsigned row_stride_elems) {
    unsigned long long ga = (unsigned long long)gptr;
    u32x4 g0 = { 1u,                                   // count=1, user descriptor
                 lds_addr,                             // LDS byte address
                 (unsigned)ga,                         // global addr [31:0]
                 (unsigned)((ga >> 32) & 0x1FFFFFFu) | (2u << 30) };  // [56:32] + type=2
    i32x8 g1 = {
        (int)(1u << 16),                               // data_size=1 -> 2 bytes
        (int)((tile_k    & 0xFFFFu) << 16),            // tensor_dim0[15:0]
        (int)((tile_rows & 0xFFFFu) << 16),            // tensor_dim0 hi = 0, tensor_dim1[15:0]
        (int)((tile_k    & 0xFFFFu) << 16),            // tensor_dim1 hi = 0, tile_dim0
        (int)( tile_rows & 0xFFFFu),                   // tile_dim1, tile_dim2=0
        (int)row_stride_elems,                         // tensor_dim0_stride [31:0]
        0, 0 };
    i32x4 gz = { 0, 0, 0, 0 };
#if __clang_major__ >= 23
    i32x8 gz8 = { 0, 0, 0, 0, 0, 0, 0, 0 };
    __builtin_amdgcn_tensor_load_to_lds(g0, g1, gz, gz, gz8, 0);
#else
    __builtin_amdgcn_tensor_load_to_lds(g0, g1, gz, gz, 0);
#endif
}
#endif  // USE_TDM

// ---------------- layout / conversion kernels ----------------

// NCHW f32 -> NHWC bf16
__global__ void k_nchw_to_nhwc_bf16(const float* __restrict__ x, unsigned short* __restrict__ out,
                                    int Bn, int C, int H, int W) {
    long long idx = (long long)blockIdx.x * blockDim.x + threadIdx.x;
    long long total = (long long)Bn * C * H * W;
    if (idx >= total) return;
    int c = (int)(idx % C);
    long long t = idx / C;
    int w = (int)(t % W); t /= W;
    int h = (int)(t % H);
    int b = (int)(t / H);
    float v = x[(((long long)b * C + c) * H + h) * W + w];
    out[idx] = f32_to_bf16(v);
}

// conv weight OIHW f32 -> Bt[N=O][K=(kh*3+kw)*Cin+c] bf16
__global__ void k_conv_w_btk(const float* __restrict__ w, unsigned short* __restrict__ Bt,
                             int O, int Cin) {
    long long idx = (long long)blockIdx.x * blockDim.x + threadIdx.x;
    long long total = (long long)O * Cin * 9;
    if (idx >= total) return;
    int c   = (int)(idx % Cin);
    int tap = (int)((idx / Cin) % 9);
    int o   = (int)(idx / ((long long)Cin * 9));
    Bt[idx] = f32_to_bf16(w[((long long)o * Cin + c) * 9 + tap]);
}

// plain f32 -> bf16
__global__ void k_cast_bf16(const float* __restrict__ s, unsigned short* __restrict__ d, long long n) {
    long long i = (long long)blockIdx.x * blockDim.x + threadIdx.x;
    if (i < n) d[i] = f32_to_bf16(s[i]);
}

// im2col for 3x3 pad-1 conv over NHWC bf16 input: A[M=B*Ho*Wo][K=9*Cin] (rows < M only)
__global__ void k_im2col3x3(const unsigned short* __restrict__ in, unsigned short* __restrict__ A,
                            int Bn, int Hin, int Win, int Cin, int Hout, int Wout, int stride) {
    int c8 = Cin >> 3;
    long long idx = (long long)blockIdx.x * blockDim.x + threadIdx.x;
    long long total = (long long)Bn * Hout * Wout * 9 * c8;
    if (idx >= total) return;
    int cblk = (int)(idx % c8);
    int tap  = (int)((idx / c8) % 9);
    long long m = idx / ((long long)c8 * 9);
    int kh = tap / 3, kw = tap % 3;
    int HW = Hout * Wout;
    int b = (int)(m / HW);
    int p = (int)(m % HW);
    int oh = p / Wout, ow = p % Wout;
    int ih = oh * stride - 1 + kh;
    int iw = ow * stride - 1 + kw;
    int K = 9 * Cin;
    uint4 val = make_uint4(0u, 0u, 0u, 0u);
    if (ih >= 0 && ih < Hin && iw >= 0 && iw < Win) {
        val = *(const uint4*)(in + (((long long)b * Hin + ih) * Win + iw) * Cin + (cblk << 3));
    }
    *(uint4*)(A + m * K + (long long)tap * Cin + (cblk << 3)) = val;
}

// ---------------- WMMA GEMM (batched, strided): C[MPADrows,N] = A[MPADrows,K] * Bt[N,K]^T + bias ----
// 256 threads = 8 waves; tile 128(M) x 64(N), k-step 32, per wave 2x2 wmma fragments.
// M is pre-padded to a multiple of 128 -> no guards, branch-free inner loop.
// Double-buffered LDS; tiles staged by the Tensor Data Mover when available.
static __device__ __forceinline__ bf16x16 frag_from_lds(const unsigned short* rowp, int kb) {
    bf16x8 lo = *(const bf16x8*)(rowp + kb);
    bf16x8 hi = *(const bf16x8*)(rowp + kb + 16);
    return __builtin_shufflevector(lo, hi, 0,1,2,3,4,5,6,7,8,9,10,11,12,13,14,15);
}

__global__ void __launch_bounds__(256)
k_gemm_bf16(const unsigned short* __restrict__ A, const unsigned short* __restrict__ Bt,
            const float* __restrict__ bias, float* __restrict__ C,
            int N, int K,
            long long sA, long long sB, long long sC,
            int biasStride, int biasDiv) {
    __shared__ __align__(16) unsigned short lA[2][128 * 32];
    __shared__ __align__(16) unsigned short lB[2][64 * 32];

    const int z = blockIdx.z;
    const int m0 = blockIdx.x * 128;
    const int n0 = blockIdx.y * 64;
    const unsigned short* Az = A + (long long)z * sA + (long long)m0 * K;
    const unsigned short* Bz = Bt + (long long)z * sB + (long long)n0 * K;
    float* Cz = C + (long long)z * sC;
    const int biasBase = (z / biasDiv) * biasStride;

    const int tid   = threadIdx.x;
    const int lane  = tid & 31;
    const int wave  = tid >> 5;
    const int wm    = wave & 3;     // M sub-block (32 rows each)
    const int wn    = wave >> 2;    // N sub-block (32 cols each)
    const int l16   = lane & 15;
    const int lhalf = lane >> 4;
    const int kb    = lhalf << 3;   // per-lane k base within 32-k tile

    f32x8 acc[2][2];
    #pragma unroll
    for (int i = 0; i < 2; ++i)
        #pragma unroll
        for (int j = 0; j < 2; ++j)
            #pragma unroll
            for (int r = 0; r < 8; ++r) acc[i][j][r] = 0.0f;

    const int kTiles = K >> 5;

#if USE_TDM
    if (wave == 0) {
        tdm_load_tile_bf16(lds_off_u32(&lA[0][0]), Az, 32u, 128u, (unsigned)K);
        tdm_load_tile_bf16(lds_off_u32(&lB[0][0]), Bz, 32u, 64u, (unsigned)K);
    }
    for (int kt = 0; kt < kTiles; ++kt) {
        const int cur = kt & 1;
        if (wave == 0) {
            if (kt + 1 < kTiles) {
                const unsigned short* An = Az + (long long)(kt + 1) * 32;
                const unsigned short* Bn = Bz + (long long)(kt + 1) * 32;
                tdm_load_tile_bf16(lds_off_u32(&lA[cur ^ 1][0]), An, 32u, 128u, (unsigned)K);
                tdm_load_tile_bf16(lds_off_u32(&lB[cur ^ 1][0]), Bn, 32u, 64u, (unsigned)K);
                __builtin_amdgcn_s_wait_tensorcnt((short)2);   // retire older (cur) pair only
            } else {
                __builtin_amdgcn_s_wait_tensorcnt((short)0);
            }
        }
        __syncthreads();   // cur tile resident in LDS for all waves

        bf16x16 af[2], bfv[2];
        #pragma unroll
        for (int i = 0; i < 2; ++i)
            af[i] = frag_from_lds(&lA[cur][(wm * 32 + i * 16 + l16) * 32], kb);
        #pragma unroll
        for (int j = 0; j < 2; ++j)
            bfv[j] = frag_from_lds(&lB[cur][(wn * 32 + j * 16 + l16) * 32], kb);
        #pragma unroll
        for (int i = 0; i < 2; ++i)
            #pragma unroll
            for (int j = 0; j < 2; ++j)
                acc[i][j] = __builtin_amdgcn_wmma_f32_16x16x32_bf16(
                    false, af[i], false, bfv[j], (short)0, acc[i][j], false, false);
        __syncthreads();   // cur fully consumed before TDM re-stages it
    }
#else
    // fallback: VGPR staging, double-buffered
    auto stage = [&](int buf, int kt) {
        const int k0 = kt << 5;
        #pragma unroll
        for (int it = 0; it < 2; ++it) {
            int chunk = tid * 2 + it;        // 0..511
            int row = chunk >> 2;
            int kq  = (chunk & 3) << 3;
            *(uint4*)(&lA[buf][row * 32 + kq]) =
                *(const uint4*)(Az + (long long)row * K + k0 + kq);
        }
        int row = tid >> 2;
        int kq  = (tid & 3) << 3;
        *(uint4*)(&lB[buf][row * 32 + kq]) =
            *(const uint4*)(Bz + (long long)row * K + k0 + kq);
    };
    stage(0, 0);
    for (int kt = 0; kt < kTiles; ++kt) {
        const int cur = kt & 1;
        __syncthreads();   // cur stores visible
        if (kt + 1 < kTiles) stage(cur ^ 1, kt + 1);

        bf16x16 af[2], bfv[2];
        #pragma unroll
        for (int i = 0; i < 2; ++i)
            af[i] = frag_from_lds(&lA[cur][(wm * 32 + i * 16 + l16) * 32], kb);
        #pragma unroll
        for (int j = 0; j < 2; ++j)
            bfv[j] = frag_from_lds(&lB[cur][(wn * 32 + j * 16 + l16) * 32], kb);
        #pragma unroll
        for (int i = 0; i < 2; ++i)
            #pragma unroll
            for (int j = 0; j < 2; ++j)
                acc[i][j] = __builtin_amdgcn_wmma_f32_16x16x32_bf16(
                    false, af[i], false, bfv[j], (short)0, acc[i][j], false, false);
        __syncthreads();
    }
#endif

    // store: lane holds rows (lhalf*8 + r), col l16 of each 16x16 tile (rows always in padded range)
    #pragma unroll
    for (int i = 0; i < 2; ++i) {
        int mbase = m0 + wm * 32 + i * 16 + (lhalf << 3);
        #pragma unroll
        for (int j = 0; j < 2; ++j) {
            int n = n0 + wn * 32 + j * 16 + l16;
            float bv = bias[biasBase + n];
            #pragma unroll
            for (int r = 0; r < 8; ++r)
                Cz[(long long)(mbase + r) * N + n] = acc[i][j][r] + bv;
        }
    }
}

// ---------------- GroupNorm + ReLU (NHWC), optional bf16 side-copy ----------------
__global__ void k_groupnorm_relu(const float* __restrict__ x, float* __restrict__ y,
                                 unsigned short* __restrict__ yb,
                                 const float* __restrict__ gamma, const float* __restrict__ beta,
                                 int P, int C, int groups) {
    int g = blockIdx.x % groups;
    int b = blockIdx.x / groups;
    int gsize = C / groups;
    int n = P * gsize;
    const float* base = x + (long long)b * P * C + g * gsize;

    float s = 0.f, s2 = 0.f;
    for (int i = threadIdx.x; i < n; i += blockDim.x) {
        int p = i / gsize, cc = i % gsize;
        float v = base[(long long)p * C + cc];
        s += v; s2 += v * v;
    }
    __shared__ float rs[256], rq[256];
    rs[threadIdx.x] = s; rq[threadIdx.x] = s2;
    __syncthreads();
    for (int st = 128; st > 0; st >>= 1) {
        if (threadIdx.x < st) { rs[threadIdx.x] += rs[threadIdx.x + st]; rq[threadIdx.x] += rq[threadIdx.x + st]; }
        __syncthreads();
    }
    float mean = rs[0] / n;
    float var  = rq[0] / n - mean * mean;
    float rstd = rsqrtf(var + 1e-5f);

    float* yo = y + (long long)b * P * C + g * gsize;
    unsigned short* bo = yb ? yb + (long long)b * P * C + g * gsize : (unsigned short*)0;
    for (int i = threadIdx.x; i < n; i += blockDim.x) {
        int p = i / gsize, cc = i % gsize;
        int ch = g * gsize + cc;
        float v = (base[(long long)p * C + cc] - mean) * rstd * gamma[ch] + beta[ch];
        v = fmaxf(v, 0.f);
        yo[(long long)p * C + cc] = v;
        if (bo) bo[(long long)p * C + cc] = f32_to_bf16(v);
    }
}

// ---------------- message passing: depthwise 5x5 per edge (reads NHWC grid slice) ----------------
__global__ void k_edge_dw(const float* __restrict__ src /*[B,49,576]*/,
                          const float* __restrict__ dw_w /*[24,64,1,5,5]*/,
                          const float* __restrict__ dw_b /*[24,64]*/,
                          unsigned short* __restrict__ out /*[24,MPAD,64] bf16*/,
                          int Bn) {
    int e = blockIdx.z;
    long long idx = (long long)blockIdx.x * blockDim.x + threadIdx.x;
    long long total = (long long)Bn * PMID * SPC;
    if (idx >= total) return;
    int c = (int)(idx % SPC);
    int p = (int)((idx / SPC) % PMID);
    int b = (int)(idx / ((long long)SPC * PMID));
    int g = c_edge_src[e];
    int oh = p / HMID, ow = p % HMID;
    float acc = dw_b[e * SPC + c];
    const float* wv = dw_w + ((long long)e * SPC + c) * 25;
    #pragma unroll
    for (int kh = 0; kh < 5; ++kh) {
        int ih = oh - 2 + kh;
        if (ih < 0 || ih >= HMID) continue;
        #pragma unroll
        for (int kw = 0; kw < 5; ++kw) {
            int iw = ow - 2 + kw;
            if (iw < 0 || iw >= HMID) continue;
            acc += src[((long long)b * PMID + ih * HMID + iw) * CMID + g * SPC + c] * wv[kh * 5 + kw];
        }
    }
    out[((long long)e * MPAD + b * PMID + p) * SPC + c] = f32_to_bf16(acc);
}

// out[b,p,g*64+c] = base[...] + sum over incoming edges of msgs[e,(b*49+p),c]
__global__ void k_aggregate(const float* __restrict__ base, const float* __restrict__ msgs,
                            float* __restrict__ out, int Bn) {
    long long idx = (long long)blockIdx.x * blockDim.x + threadIdx.x;
    long long total = (long long)Bn * PMID * CMID;
    if (idx >= total) return;
    int ch = (int)(idx % CMID);
    long long m = idx / CMID;
    int g = ch / SPC, c = ch % SPC;
    float acc = base[idx];
    for (int e = c_node_off[g]; e < c_node_off[g + 1]; ++e)
        acc += msgs[((long long)e * MPAD + m) * SPC + c];
    out[idx] = acc;
}

// ---------------- grouped ConvTranspose2d (k4 s2 p1, groups=9) as 36 batched GEMMs ----------------
// A[z=(g,cls)][m=(b,7x7)][k=(tap,ic)] bf16, built from NHWC fp32 src [B,7,7,576]
__global__ void k_dec1_im2col(const float* __restrict__ src, unsigned short* __restrict__ A, int Bn) {
    long long idx = (long long)blockIdx.x * blockDim.x + threadIdx.x;
    long long total = (long long)36 * Bn * PMID * 4 * 8;   // 8 chunks of 8 ic
    if (idx >= total) return;
    int chunk = (int)(idx % 8);
    int t = (int)((idx / 8) % 4);
    long long m = (idx / 32) % ((long long)Bn * PMID);
    int z = (int)(idx / (32LL * Bn * PMID));
    int g = z >> 2, cls = z & 3;
    int r = cls >> 1, s = cls & 1;
    int khi = t >> 1, kwi = t & 1;
    int b = (int)(m / PMID), p = (int)(m % PMID);
    int ohh = p / HMID, oww = p % HMID;
    int ih = ohh + r - khi;
    int iw = oww + s - kwi;
    unsigned short h[8];
    if (ih >= 0 && ih < HMID && iw >= 0 && iw < HMID) {
        const float* sp = src + ((long long)b * PMID + ih * HMID + iw) * CMID + g * SPC + chunk * 8;
        #pragma unroll
        for (int i = 0; i < 8; ++i) h[i] = f32_to_bf16(sp[i]);
    } else {
        #pragma unroll
        for (int i = 0; i < 8; ++i) h[i] = 0;
    }
    uint4 v;
    v.x = (unsigned)h[0] | ((unsigned)h[1] << 16);
    v.y = (unsigned)h[2] | ((unsigned)h[3] << 16);
    v.z = (unsigned)h[4] | ((unsigned)h[5] << 16);
    v.w = (unsigned)h[6] | ((unsigned)h[7] << 16);
    *(uint4*)(A + (((long long)z * MPAD + m) * 256) + t * 64 + chunk * 8) = v;
}

// Bt[z=(g,cls)][oc][k=(tap,ic)] from up1_w [576, 64, 4, 4]
__global__ void k_dec1_w(const float* __restrict__ w, unsigned short* __restrict__ Bt) {
    long long idx = (long long)blockIdx.x * blockDim.x + threadIdx.x;
    long long total = 36LL * 64 * 256;
    if (idx >= total) return;
    int k  = (int)(idx % 256);
    int oc = (int)((idx / 256) % 64);
    int z  = (int)(idx / (256 * 64));
    int g = z >> 2, cls = z & 3;
    int r = cls >> 1, s = cls & 1;
    int t = k / 64, ic = k % 64;
    int kh = 1 - r + 2 * (t >> 1);
    int kw = 1 - s + 2 * (t & 1);
    Bt[idx] = f32_to_bf16(w[(((long long)g * 64 + ic) * 64 + oc) * 16 + kh * 4 + kw]);
}

// scatter GEMM result [36][MPAD][64] -> NHWC [B,14,14,576]
__global__ void k_dec1_scatter(const float* __restrict__ Cd, float* __restrict__ t1, int Bn) {
    long long idx = (long long)blockIdx.x * blockDim.x + threadIdx.x;
    long long total = 36LL * Bn * PMID * 64;
    if (idx >= total) return;
    int oc = (int)(idx % 64);
    long long m = (idx / 64) % ((long long)Bn * PMID);
    int z = (int)(idx / (64LL * Bn * PMID));
    int g = z >> 2, cls = z & 3;
    int r = cls >> 1, s = cls & 1;
    int b = (int)(m / PMID), p = (int)(m % PMID);
    int oh = 2 * (p / HMID) + r;
    int ow = 2 * (p % HMID) + s;
    t1[(((long long)b * 196) + oh * 14 + ow) * CMID + g * SPC + oc] =
        Cd[((long long)z * MPAD + m) * 64 + oc];
}

// final ConvTranspose2d (576->9, k4 s2 p1, groups=9), writes NCHW output slice [B,9,28,28]
__global__ void k_up2(const float* __restrict__ t1 /*[B,196,576]*/, const float* __restrict__ w /*[576,1,4,4]*/,
                      const float* __restrict__ bias /*[9]*/, float* __restrict__ out, int Bn) {
    long long idx = (long long)blockIdx.x * blockDim.x + threadIdx.x;
    long long total = (long long)Bn * NGRID * 28 * 28;
    if (idx >= total) return;
    int ow = (int)(idx % 28);
    int oh = (int)((idx / 28) % 28);
    int g  = (int)((idx / 784) % NGRID);
    int b  = (int)(idx / (784 * NGRID));
    float acc = bias[g];
    #pragma unroll
    for (int kh = 0; kh < 4; ++kh) {
        int nh = oh + 1 - kh;
        if (nh < 0 || (nh & 1)) continue;
        int ih = nh >> 1;
        if (ih >= 14) continue;
        #pragma unroll
        for (int kw = 0; kw < 4; ++kw) {
            int nw = ow + 1 - kw;
            if (nw < 0 || (nw & 1)) continue;
            int iw = nw >> 1;
            if (iw >= 14) continue;
            const float* sp = t1 + ((long long)b * 196 + ih * 14 + iw) * CMID + g * SPC;
            const float* wp = w + (long long)g * SPC * 16 + kh * 4 + kw;
            for (int ic = 0; ic < SPC; ++ic)
                acc += sp[ic] * wp[ic * 16];
        }
    }
    out[idx] = acc;
}

// ---------------- host orchestration ----------------
static inline unsigned cdiv_u(long long a, int b) { return (unsigned)((a + b - 1) / b); }

extern "C" void kernel_launch(void* const* d_in, const int* in_sizes, int n_in,
                              void* d_out, int out_size, void* d_ws, size_t ws_size,
                              hipStream_t stream) {
    (void)in_sizes; (void)n_in; (void)out_size; (void)ws_size;
    const float* x        = (const float*)d_in[0];
    const float* w0       = (const float*)d_in[1];
    const float* b0       = (const float*)d_in[2];
    const float* wc       = (const float*)d_in[3];
    const float* bc       = (const float*)d_in[4];
    const float* gng      = (const float*)d_in[5];
    const float* gnbta    = (const float*)d_in[6];
    const float* fo_dw_w  = (const float*)d_in[7];
    const float* fo_dw_b  = (const float*)d_in[8];
    const float* fo_pw_w  = (const float*)d_in[9];
    const float* fo_pw_b  = (const float*)d_in[10];
    const float* so_dw_w  = (const float*)d_in[11];
    const float* so_dw_b  = (const float*)d_in[12];
    const float* so_pw_w  = (const float*)d_in[13];
    const float* so_pw_b  = (const float*)d_in[14];
    const float* up1_w    = (const float*)d_in[15];
    const float* up1_b    = (const float*)d_in[16];
    const float* sbn_g    = (const float*)d_in[17];
    const float* sbn_b    = (const float*)d_in[18];
    const float* up2_w    = (const float*)d_in[19];
    const float* up2_b    = (const float*)d_in[20];
    float* out = (float*)d_out;

    char* ws = (char*)d_ws;
    size_t off = 0;
    auto take = [&](size_t bytes) -> void* {
        void* p = ws + off;
        off = (off + bytes + 255) & ~(size_t)255;
        return p;
    };
    unsigned short* xb   = (unsigned short*)take((size_t)BATCH * HIN * HIN * CIN * 2);
    unsigned short* wbuf = (unsigned short*)take((size_t)CMID * 5184 * 2);
    unsigned short* Abuf = (unsigned short*)take((size_t)36 * MPAD * 256 * 2);   // >= MPAD*5184*2
    float* ybuf          = (float*)take((size_t)36 * MPAD * 64 * 4);             // >= MPAD*576*4
    float* hf0           = (float*)take((size_t)M_MID * CMID * 4);
    float* hf1           = (float*)take((size_t)M_MID * CMID * 4);
    unsigned short* hb0  = (unsigned short*)take((size_t)M_MID * CMID * 2);
    unsigned short* hb1  = (unsigned short*)take((size_t)M_MID * CMID * 2);
    unsigned short* dwb  = (unsigned short*)take((size_t)NEDGE * MPAD * SPC * 2);
    float* msgs          = (float*)take((size_t)NEDGE * MPAD * SPC * 4);
    float* firstb        = (float*)take((size_t)M_MID * CMID * 4);
    float* x2b           = (float*)take((size_t)M_MID * CMID * 4);
    unsigned short* pwb  = (unsigned short*)take((size_t)NEDGE * SPC * SPC * 2);
    unsigned short* decw = (unsigned short*)take((size_t)36 * 64 * 256 * 2);
    float* t1            = (float*)take((size_t)BATCH * 196 * CMID * 4);

    // ---- input to NHWC bf16
    k_nchw_to_nhwc_bf16<<<cdiv_u((long long)BATCH * CIN * HIN * HIN, 256), 256, 0, stream>>>(
        x, xb, BATCH, CIN, HIN, HIN);

    // ---- conv0 (stride 2) + GN + ReLU
    k_conv_w_btk<<<cdiv_u((long long)CMID * CIN * 9, 256), 256, 0, stream>>>(w0, wbuf, CMID, CIN);
    k_im2col3x3<<<cdiv_u((long long)M_MID * 9 * (CIN / 8), 256), 256, 0, stream>>>(
        xb, Abuf, BATCH, HIN, HIN, CIN, HMID, HMID, 2);
    k_gemm_bf16<<<dim3(MPAD / 128, CMID / 64, 1), 256, 0, stream>>>(
        Abuf, wbuf, b0, ybuf, CMID, 9 * CIN, 0, 0, 0, 0, 1);
    k_groupnorm_relu<<<BATCH * 36, 256, 0, stream>>>(ybuf, hf0, hb0, gng, gnbta, PMID, CMID, 36);

    // ---- 7 stride-1 convs + GN + ReLU
    float* hf_cur = hf0; float* hf_nxt = hf1;
    unsigned short* hb_cur = hb0; unsigned short* hb_nxt = hb1;
    for (int i = 0; i < 7; ++i) {
        k_conv_w_btk<<<cdiv_u((long long)CMID * CMID * 9, 256), 256, 0, stream>>>(
            wc + (long long)i * CMID * CMID * 9, wbuf, CMID, CMID);
        k_im2col3x3<<<cdiv_u((long long)M_MID * 9 * (CMID / 8), 256), 256, 0, stream>>>(
            hb_cur, Abuf, BATCH, HMID, HMID, CMID, HMID, HMID, 1);
        k_gemm_bf16<<<dim3(MPAD / 128, CMID / 64, 1), 256, 0, stream>>>(
            Abuf, wbuf, bc + (long long)i * CMID, ybuf, CMID, 9 * CMID, 0, 0, 0, 0, 1);
        k_groupnorm_relu<<<BATCH * 36, 256, 0, stream>>>(
            ybuf, hf_nxt, hb_nxt, gng + (long long)(i + 1) * CMID, gnbta + (long long)(i + 1) * CMID,
            PMID, CMID, 36);
        float* tf = hf_cur; hf_cur = hf_nxt; hf_nxt = tf;
        unsigned short* tb = hb_cur; hb_cur = hb_nxt; hb_nxt = tb;
    }
    float* hfin = hf_cur;  // conv stack output [B,49,576]

    // ---- first-order message passing
    k_edge_dw<<<dim3(cdiv_u((long long)BATCH * PMID * SPC, 256), 1, NEDGE), 256, 0, stream>>>(
        hfin, fo_dw_w, fo_dw_b, dwb, BATCH);
    k_cast_bf16<<<cdiv_u((long long)NEDGE * SPC * SPC, 256), 256, 0, stream>>>(
        fo_pw_w, pwb, (long long)NEDGE * SPC * SPC);
    k_gemm_bf16<<<dim3(MPAD / 128, 1, NEDGE), 256, 0, stream>>>(
        dwb, pwb, fo_pw_b, msgs, SPC, SPC,
        (long long)MPAD * SPC, (long long)SPC * SPC, (long long)MPAD * SPC, SPC, 1);
    k_aggregate<<<cdiv_u((long long)M_MID * CMID, 256), 256, 0, stream>>>(hfin, msgs, firstb, BATCH);

    // ---- second-order message passing
    k_edge_dw<<<dim3(cdiv_u((long long)BATCH * PMID * SPC, 256), 1, NEDGE), 256, 0, stream>>>(
        firstb, so_dw_w, so_dw_b, dwb, BATCH);
    k_cast_bf16<<<cdiv_u((long long)NEDGE * SPC * SPC, 256), 256, 0, stream>>>(
        so_pw_w, pwb, (long long)NEDGE * SPC * SPC);
    k_gemm_bf16<<<dim3(MPAD / 128, 1, NEDGE), 256, 0, stream>>>(
        dwb, pwb, so_pw_b, msgs, SPC, SPC,
        (long long)MPAD * SPC, (long long)SPC * SPC, (long long)MPAD * SPC, SPC, 1);
    k_aggregate<<<cdiv_u((long long)M_MID * CMID, 256), 256, 0, stream>>>(hfin, msgs, x2b, BATCH);

    // ---- heads
    k_dec1_w<<<cdiv_u(36LL * 64 * 256, 256), 256, 0, stream>>>(up1_w, decw);
    const long long HEAD_OUT = (long long)BATCH * NGRID * 28 * 28;
    const float* head_src[2] = { hfin, x2b };
    for (int hid = 0; hid < 2; ++hid) {
        k_dec1_im2col<<<cdiv_u(36LL * BATCH * PMID * 32, 256), 256, 0, stream>>>(
            head_src[hid], Abuf, BATCH);
        k_gemm_bf16<<<dim3(MPAD / 128, 1, 36), 256, 0, stream>>>(
            Abuf, decw, up1_b, ybuf, 64, 256,
            (long long)MPAD * 256, 64LL * 256, (long long)MPAD * 64, 64, 4);
        k_dec1_scatter<<<cdiv_u(36LL * BATCH * PMID * 64, 256), 256, 0, stream>>>(ybuf, t1, BATCH);
        k_groupnorm_relu<<<BATCH * NGRID, 256, 0, stream>>>(
            t1, t1, (unsigned short*)0, sbn_g, sbn_b, 196, CMID, NGRID);
        k_up2<<<cdiv_u(HEAD_OUT, 256), 256, 0, stream>>>(
            t1, up2_w, up2_b, out + hid * HEAD_OUT, BATCH);
    }
}
